// TopologicallyRegularizedAutoencoder_1906965479929
// MI455X (gfx1250) — compile-verified, matrix operand order checked by reading, not run on hardware
//
#include <hip/hip_runtime.h>
#include <hip/hip_bf16.h>

typedef __attribute__((ext_vector_type(16))) _Float16 v16h;
typedef __attribute__((ext_vector_type(8)))  _Float16 v8h;
typedef __attribute__((ext_vector_type(8)))  float    v8f;

#define NB   512      // batch
#define DIM  256      // input dim
#define LAMB 0.43f

// ---------------------------------------------------------------------------
// helpers
// ---------------------------------------------------------------------------
__device__ inline v16h load_frag_rowmajor(const _Float16* P, int ld, int row0, int k0, int lane) {
  // 16-bit A/B fragment layout (CDNA5 ISA 7.12.2):
  //  lanes 0-15 : row = lane,    halves 0-7 = K k0..k0+7,  halves 8-15 = K k0+16..k0+23
  //  lanes 16-31: row = lane-16, halves 0-7 = K k0+8..k0+15, halves 8-15 = K k0+24..k0+31
  int r  = row0 + (lane & 15);
  int kk = k0 + ((lane >> 4) << 3);
  const _Float16* p = P + (size_t)r * ld + kk;
  v8h lo = *(const v8h*)(p);
  v8h hi = *(const v8h*)(p + 16);
  v16h f;
#pragma unroll
  for (int i = 0; i < 8; ++i) { f[i] = lo[i]; f[i + 8] = hi[i]; }
  return f;
}

// ---------------------------------------------------------------------------
// conversion kernels
// ---------------------------------------------------------------------------
__global__ void conv_x_kernel(const float* __restrict__ x, _Float16* __restrict__ xh,
                              float* __restrict__ nx) {
  int row = blockIdx.x;          // 512 rows
  int t   = threadIdx.x;         // 256 threads == DIM
  float v = x[row * DIM + t];
  xh[row * DIM + t] = (_Float16)v;
  __shared__ float sm[256];
  sm[t] = v * v;
  __syncthreads();
  for (int off = 128; off > 0; off >>= 1) {
    if (t < off) sm[t] += sm[t + off];
    __syncthreads();
  }
  if (t == 0) nx[row] = sm[0];
}

__global__ void transpose_conv_kernel(const float* __restrict__ W, _Float16* __restrict__ Wt,
                                      int din, int dout) {
  int i = blockIdx.x * blockDim.x + threadIdx.x;
  if (i >= din * dout) return;
  int r = i / dout, c = i % dout;                 // W[r][c], row-major [din,dout]
  Wt[(size_t)c * din + r] = (_Float16)W[i];       // Wt row-major [dout,din]
}

// ---------------------------------------------------------------------------
// WMMA GEMM:  C[M,N] = A[M,K] * Bt[N,K]^T  (+bias) (relu?)  -> f16 and/or f32
// one 16x16 tile per wave, 8 waves / block
// ---------------------------------------------------------------------------
__global__ void gemm_wmma_kernel(const _Float16* __restrict__ A, const _Float16* __restrict__ Bt,
                                 const float* __restrict__ bias,
                                 _Float16* __restrict__ outH, float* __restrict__ outF,
                                 int M, int N, int K, int relu) {
  int lane   = threadIdx.x & 31;
  int wave   = threadIdx.x >> 5;
  int tilesN = N >> 4;
  int tile   = blockIdx.x * 8 + wave;
  int tm     = tile / tilesN;
  int tn     = tile % tilesN;
  if (tm * 16 >= M) return;                        // wave-uniform

  v8f c = {};
  for (int k0 = 0; k0 < K; k0 += 32) {
    v16h a = load_frag_rowmajor(A,  K, tm * 16, k0, lane);
    v16h b = load_frag_rowmajor(Bt, K, tn * 16, k0, lane);
    c = __builtin_amdgcn_wmma_f32_16x16x32_f16(false, a, false, b, (short)0, c, false, false);
  }

  int   n  = tn * 16 + (lane & 15);
  float bv = bias ? bias[n] : 0.f;
  int   mb = tm * 16 + ((lane >> 4) << 3);         // +8 rows for hi half-wave
#pragma unroll
  for (int r = 0; r < 8; ++r) {
    int   m = mb + r;
    float v = c[r] + bv;
    if (relu) v = fmaxf(v, 0.f);
    if (outH) outH[(size_t)m * N + n] = (_Float16)v;
    if (outF) outF[(size_t)m * N + n] = v;
  }
}

// ---------------------------------------------------------------------------
// tiny GEMMs around the 2-D bottleneck
// ---------------------------------------------------------------------------
__global__ void latent_kernel(const _Float16* __restrict__ h2, const float* __restrict__ Wm,
                              const float* __restrict__ bm, float* __restrict__ lat) {
  int m = blockIdx.x * blockDim.x + threadIdx.x;
  if (m >= NB) return;
  float a0 = bm[0], a1 = bm[1];
  for (int k = 0; k < 64; ++k) {
    float h = (float)h2[m * 64 + k];
    a0 += h * Wm[k * 2 + 0];
    a1 += h * Wm[k * 2 + 1];
  }
  lat[m * 2 + 0] = a0;
  lat[m * 2 + 1] = a1;
}

__global__ void h3_kernel(const float* __restrict__ lat, const float* __restrict__ W3,
                          const float* __restrict__ b3, _Float16* __restrict__ h3) {
  int i = blockIdx.x * blockDim.x + threadIdx.x;   // NB*64
  if (i >= NB * 64) return;
  int m = i >> 6, n = i & 63;
  float v = b3[n] + lat[m * 2] * W3[n] + lat[m * 2 + 1] * W3[64 + n];
  h3[i] = (_Float16)fmaxf(v, 0.f);
}

// ---------------------------------------------------------------------------
// distance matrices
// ---------------------------------------------------------------------------
__global__ void xd_kernel(const float* __restrict__ G, const float* __restrict__ nx,
                          float* __restrict__ xd, unsigned int* __restrict__ maxu) {
  int i = blockIdx.x * blockDim.x + threadIdx.x;   // NB*NB
  int r = i >> 9, c = i & (NB - 1);
  float d2 = nx[r] + nx[c] - 2.f * G[i];
  float d  = (r == c) ? 0.f : sqrtf(fmaxf(d2, 0.f));
  xd[i] = d;
  __shared__ float sm[256];
  sm[threadIdx.x] = d;
  __syncthreads();
  for (int off = 128; off > 0; off >>= 1) {
    if ((int)threadIdx.x < off) sm[threadIdx.x] = fmaxf(sm[threadIdx.x], sm[threadIdx.x + off]);
    __syncthreads();
  }
  if (threadIdx.x == 0) atomicMax(maxu, __float_as_uint(sm[0]));   // nonneg: uint order == float order
}

__global__ void xd_norm_kernel(float* __restrict__ xd, const unsigned int* __restrict__ maxu) {
  float m   = __uint_as_float(*maxu);
  float inv = (m > 0.f) ? 1.f / m : 0.f;
  int   i   = blockIdx.x * blockDim.x + threadIdx.x;
  xd[i] *= inv;
}

__global__ void ld_kernel(const float* __restrict__ lat, const float* __restrict__ lnorm,
                          float* __restrict__ ldm) {
  int i = blockIdx.x * blockDim.x + threadIdx.x;   // NB*NB
  int r = i >> 9, c = i & (NB - 1);
  float dx = lat[r * 2] - lat[c * 2];
  float dy = lat[r * 2 + 1] - lat[c * 2 + 1];
  ldm[i] = sqrtf(dx * dx + dy * dy) / lnorm[0];
}

// ---------------------------------------------------------------------------
// reconstruction loss
// ---------------------------------------------------------------------------
__global__ void ae_kernel(const float* __restrict__ x, const float* __restrict__ xr,
                          float* __restrict__ ae) {
  int i = blockIdx.x * blockDim.x + threadIdx.x;   // NB*DIM
  float d = x[i] - xr[i];
  __shared__ float sm[256];
  sm[threadIdx.x] = d * d;
  __syncthreads();
  for (int off = 128; off > 0; off >>= 1) {
    if ((int)threadIdx.x < off) sm[threadIdx.x] += sm[threadIdx.x + off];
    __syncthreads();
  }
  if (threadIdx.x == 0) atomicAdd(ae, sm[0]);
}

// ---------------------------------------------------------------------------
// Boruvka MST (0-dim persistence pairs == MST edges for distinct weights)
// ---------------------------------------------------------------------------
__global__ void mst_init_kernel(int* __restrict__ comp) {
  int t = blockIdx.x * blockDim.x + threadIdx.x;
  if (t < NB) comp[t] = t;
}

__global__ void boruvka_reset_kernel(unsigned long long* __restrict__ minE, int* __restrict__ link) {
  int t = blockIdx.x * blockDim.x + threadIdx.x;
  if (t < NB) { minE[t] = ~0ull; link[t] = t; }
}

__global__ void boruvka_find_kernel(const float* __restrict__ D, const int* __restrict__ comp,
                                    unsigned long long* __restrict__ minE) {
  int u = blockIdx.x * blockDim.x + threadIdx.x;
  if (u >= NB) return;
  int cu = comp[u];
  unsigned long long best = ~0ull;
  for (int v = 0; v < NB; ++v) {
    if (comp[v] == cu) continue;
    float w = D[u * NB + v];
    unsigned long long key =
        ((unsigned long long)__float_as_uint(w) << 32) | (unsigned)((u << 16) | v);
    if (key < best) best = key;
  }
  if (best != ~0ull) atomicMin(&minE[cu], best);
}

__global__ void boruvka_join_kernel(const unsigned long long* __restrict__ minE,
                                    const int* __restrict__ comp, int* __restrict__ link,
                                    int2* __restrict__ edges, int* __restrict__ count) {
  int a = blockIdx.x * blockDim.x + threadIdx.x;   // component id == root vertex id
  if (a >= NB) return;
  unsigned long long k = minE[a];
  if (k == ~0ull) return;
  unsigned code = (unsigned)k;
  int u = (int)(code >> 16), v = (int)(code & 0xffff);
  int b = comp[v];
  if (b == a) return;
  // mutual (2-cycle) dedup: if b's min edge points back into a, only smaller root acts
  unsigned long long kb = minE[b];
  if (kb != ~0ull) {
    int vb = (int)((unsigned)kb & 0xffff);
    if (comp[vb] == a && a > b) return;
  }
  link[a] = b;                                     // hook a -> b
  int idx = atomicAdd(count, 1);
  if (idx < NB) edges[idx] = make_int2(min(u, v), max(u, v));
}

__global__ void boruvka_flatten_kernel(int* __restrict__ link, int* __restrict__ comp) {
  int t = threadIdx.x;                              // <<<1,512>>>
  for (int it = 0; it < 10; ++it) {
    int l = link[link[t]];
    __syncthreads();
    link[t] = l;
    __syncthreads();
  }
  comp[t] = link[comp[t]];
}

// ---------------------------------------------------------------------------
// topo + final loss
// ---------------------------------------------------------------------------
__global__ void topo_kernel(const int2* __restrict__ e1, const int* __restrict__ c1,
                            const int2* __restrict__ e2, const int* __restrict__ c2,
                            const float* __restrict__ xd, const float* __restrict__ ld,
                            float* __restrict__ topo) {
  int i  = blockIdx.x * blockDim.x + threadIdx.x;
  int n1 = min(c1[0], NB), n2 = min(c2[0], NB);
  float s = 0.f;
  if (i < n1) { int2 e = e1[i]; float a = xd[e.x * NB + e.y] - ld[e.x * NB + e.y]; s += a * a; }
  if (i < n2) { int2 e = e2[i]; float a = ld[e.x * NB + e.y] - xd[e.x * NB + e.y]; s += a * a; }
  if (s != 0.f) atomicAdd(topo, s);
}

__global__ void init_scalars_kernel(float* ae, float* topo, unsigned int* maxu, int* c1, int* c2) {
  if (threadIdx.x == 0) { *ae = 0.f; *topo = 0.f; *maxu = 0u; *c1 = 0; *c2 = 0; }
}

__global__ void final_kernel(const float* __restrict__ ae, const float* __restrict__ topo,
                             float* __restrict__ out) {
  if (threadIdx.x == 0)
    out[0] = ae[0] * (1.f / (float)(NB * DIM)) + LAMB * topo[0] * (1.f / (float)NB);
}

// ---------------------------------------------------------------------------
// launch
// ---------------------------------------------------------------------------
extern "C" void kernel_launch(void* const* d_in, const int* in_sizes, int n_in,
                              void* d_out, int out_size, void* d_ws, size_t ws_size,
                              hipStream_t stream) {
  const float* x  = (const float*)d_in[0];
  const float* W1 = (const float*)d_in[2];  const float* b1 = (const float*)d_in[3];
  const float* W2 = (const float*)d_in[4];  const float* b2 = (const float*)d_in[5];
  const float* Wm = (const float*)d_in[6];  const float* bm = (const float*)d_in[7];
  const float* W3 = (const float*)d_in[8];  const float* b3 = (const float*)d_in[9];
  const float* W4 = (const float*)d_in[10]; const float* b4 = (const float*)d_in[11];
  const float* W5 = (const float*)d_in[12]; const float* b5 = (const float*)d_in[13];
  const float* lnorm = (const float*)d_in[14];
  float* out = (float*)d_out;

  // workspace layout (256B aligned slabs)
  char*  ws = (char*)d_ws;
  size_t o  = 0;
  auto alloc = [&](size_t bytes) { char* p = ws + o; o = (o + bytes + 255) & ~(size_t)255; return p; };
  _Float16* xh  = (_Float16*)alloc(NB * DIM * 2);
  _Float16* W1t = (_Float16*)alloc(256 * 128 * 2);
  _Float16* W2t = (_Float16*)alloc(128 * 64 * 2);
  _Float16* W4t = (_Float16*)alloc(64 * 128 * 2);
  _Float16* W5t = (_Float16*)alloc(128 * 256 * 2);
  _Float16* h1  = (_Float16*)alloc(NB * 128 * 2);
  _Float16* h2  = (_Float16*)alloc(NB * 64 * 2);
  _Float16* h3  = (_Float16*)alloc(NB * 64 * 2);
  _Float16* h4  = (_Float16*)alloc(NB * 128 * 2);
  float*    lat = (float*)alloc(NB * 2 * 4);
  float*    xr  = (float*)alloc(NB * DIM * 4);
  float*    G   = (float*)alloc(NB * NB * 4);
  float*    xd  = (float*)alloc(NB * NB * 4);
  float*    ld  = (float*)alloc(NB * NB * 4);
  float*    nx  = (float*)alloc(NB * 4);
  float*        ae    = (float*)alloc(4);
  float*        topo  = (float*)alloc(4);
  unsigned int* maxu  = (unsigned int*)alloc(4);
  int*          cnt1  = (int*)alloc(4);
  int*          cnt2  = (int*)alloc(4);
  int*  comp1 = (int*)alloc(NB * 4);  int* link1 = (int*)alloc(NB * 4);
  unsigned long long* minE1 = (unsigned long long*)alloc(NB * 8);
  int2* edg1  = (int2*)alloc(NB * 8);
  int*  comp2 = (int*)alloc(NB * 4);  int* link2 = (int*)alloc(NB * 4);
  unsigned long long* minE2 = (unsigned long long*)alloc(NB * 8);
  int2* edg2  = (int2*)alloc(NB * 8);

  init_scalars_kernel<<<1, 64, 0, stream>>>(ae, topo, maxu, cnt1, cnt2);

  // f16 staging: x (with row sq-norms) and transposed weights
  conv_x_kernel<<<NB, 256, 0, stream>>>(x, xh, nx);
  transpose_conv_kernel<<<(256 * 128 + 255) / 256, 256, 0, stream>>>(W1, W1t, 256, 128);
  transpose_conv_kernel<<<(128 * 64 + 255) / 256, 256, 0, stream>>>(W2, W2t, 128, 64);
  transpose_conv_kernel<<<(64 * 128 + 255) / 256, 256, 0, stream>>>(W4, W4t, 64, 128);
  transpose_conv_kernel<<<(128 * 256 + 255) / 256, 256, 0, stream>>>(W5, W5t, 128, 256);

  // encoder / decoder WMMA GEMMs (tiles = M/16 * N/16, 8 waves per block)
  gemm_wmma_kernel<<<(NB / 16) * (128 / 16) / 8, 256, 0, stream>>>(xh, W1t, b1, h1, nullptr, NB, 128, 256, 1);
  gemm_wmma_kernel<<<(NB / 16) * (64 / 16) / 8, 256, 0, stream>>>(h1, W2t, b2, h2, nullptr, NB, 64, 128, 1);
  latent_kernel<<<2, 256, 0, stream>>>(h2, Wm, bm, lat);
  h3_kernel<<<(NB * 64) / 256, 256, 0, stream>>>(lat, W3, b3, h3);
  gemm_wmma_kernel<<<(NB / 16) * (128 / 16) / 8, 256, 0, stream>>>(h3, W4t, b4, h4, nullptr, NB, 128, 64, 1);
  gemm_wmma_kernel<<<(NB / 16) * (256 / 16) / 8, 256, 0, stream>>>(h4, W5t, b5, nullptr, xr, NB, 256, 128, 0);

  // Gram matrix X·X^T via WMMA (Bt = X itself), then distances
  gemm_wmma_kernel<<<(NB / 16) * (NB / 16) / 8, 256, 0, stream>>>(xh, xh, nullptr, nullptr, G, NB, NB, 256, 0);
  xd_kernel<<<(NB * NB) / 256, 256, 0, stream>>>(G, nx, xd, maxu);
  xd_norm_kernel<<<(NB * NB) / 256, 256, 0, stream>>>(xd, maxu);
  ld_kernel<<<(NB * NB) / 256, 256, 0, stream>>>(lat, lnorm, ld);

  ae_kernel<<<(NB * DIM) / 256, 256, 0, stream>>>(x, xr, ae);

  // MST via Boruvka on both distance matrices (10 rounds >= log2(512))
  const float* Ds[2]  = { xd, ld };
  int*  comps[2] = { comp1, comp2 };
  int*  links[2] = { link1, link2 };
  unsigned long long* mins[2] = { minE1, minE2 };
  int2* edges[2] = { edg1, edg2 };
  int*  cnts[2]  = { cnt1, cnt2 };
  for (int m = 0; m < 2; ++m) {
    mst_init_kernel<<<2, 256, 0, stream>>>(comps[m]);
    for (int r = 0; r < 10; ++r) {
      boruvka_reset_kernel<<<2, 256, 0, stream>>>(mins[m], links[m]);
      boruvka_find_kernel<<<2, 256, 0, stream>>>(Ds[m], comps[m], mins[m]);
      boruvka_join_kernel<<<2, 256, 0, stream>>>(mins[m], comps[m], links[m], edges[m], cnts[m]);
      boruvka_flatten_kernel<<<1, 512, 0, stream>>>(links[m], comps[m]);
    }
  }

  topo_kernel<<<2, 256, 0, stream>>>(edg1, cnt1, edg2, cnt2, xd, ld, topo);
  final_kernel<<<1, 32, 0, stream>>>(ae, topo, out);
  (void)in_sizes; (void)n_in; (void)out_size; (void)ws_size;
}
